// LLM_2637109919994
// MI455X (gfx1250) — matrix operand order repaired and checked by reading, not compiled, not run on hardware
//
#include <hip/hip_runtime.h>
#include <hip/hip_bf16.h>

// ---------------------------------------------------------------------------
// GPT forward (B=2,T=1024,V=32000,D=768,H=12,L=4,FF=3072) for gfx1250.
// All GEMMs run through v_wmma_f32_16x16x32_bf16 (wave32), double-buffered
// LDS pipeline using GLOBAL_LOAD_ASYNC_TO_LDS_B128 (ASYNCcnt /
// s_wait_asynccnt) when available, register pipelining otherwise. All weights
// are converted fp32 -> bf16 *transposed* ([N][K]) each launch so every
// B-operand is K-contiguous (16B vector copies only, no LDS scatter).
// ---------------------------------------------------------------------------

typedef __attribute__((ext_vector_type(16))) __bf16 v16bf;
typedef __attribute__((ext_vector_type(8)))  __bf16 v8bf;
typedef __attribute__((ext_vector_type(8)))  float  v8f;
typedef __attribute__((ext_vector_type(8)))  unsigned short v8us;
typedef int v4i_g __attribute__((vector_size(16)));   // matches builtin proto

#define CB 2
#define CT 1024
#define CV 32000
#define CD 768
#define CH 12
#define CL 4
#define CDH 64
#define CFF 3072
#define CM (CB*CT)          // 2048 token rows

#if defined(__has_builtin)
#if __has_builtin(__builtin_amdgcn_global_load_async_to_lds_b128)
#define USE_ASYNC_LDS 1
#endif
#endif

__device__ __forceinline__ unsigned short f32_to_bf16_bits(float f) {
  unsigned int u = __float_as_uint(f);
  u += 0x7FFFu + ((u >> 16) & 1u);   // round to nearest even
  return (unsigned short)(u >> 16);
}
__device__ __forceinline__ float gelu_exact(float x) {
  return 0.5f * x * (1.0f + erff(x * 0.70710678118654752f));
}

#ifdef USE_ASYNC_LDS
// 16B global -> LDS DMA (tracked with ASYNCcnt)
__device__ __forceinline__ void cp16_async(const unsigned short* g,
                                           unsigned short* l) {
  __builtin_amdgcn_global_load_async_to_lds_b128(
      (v4i_g __attribute__((address_space(1)))*)(unsigned long long)g,
      (v4i_g __attribute__((address_space(3)))*)(unsigned)(unsigned long long)l,
      0, 0);
}
__device__ __forceinline__ void wait_async0() {
#if __has_builtin(__builtin_amdgcn_s_wait_asynccnt)
  __builtin_amdgcn_s_wait_asynccnt(0);
#else
  asm volatile("s_wait_asynccnt 0x0" ::: "memory");
#endif
}
#endif

// --------------------- fp32 -> bf16 transposed convert ----------------------
// s: [nmat][Kd][Nd] fp32  ->  d: [nmat][Nd][Kd] bf16
__global__ __launch_bounds__(256) void k_cvt_bf16_T(const float* __restrict__ s,
                                                    unsigned short* __restrict__ d,
                                                    int Kd, int Nd, long long total) {
  const long long KN = (long long)Kd * Nd;
  long long i = (long long)blockIdx.x * 256 + threadIdx.x;
  const long long stride = (long long)gridDim.x * 256;
  for (; i < total; i += stride) {
    long long m = i / KN;
    long long r = i - m * KN;
    int k = (int)(r / Nd);
    int n = (int)(r - (long long)k * Nd);
    d[m * KN + (long long)n * Kd + k] = f32_to_bf16_bits(s[i]);
  }
}

// ------------------------------- embedding ----------------------------------
__global__ __launch_bounds__(256) void k_embed(const int* __restrict__ x,
                                               const float* __restrict__ tok,
                                               const float* __restrict__ pos,
                                               float* __restrict__ h,
                                               long long total) {
  long long i = (long long)blockIdx.x * 256 + threadIdx.x;
  const long long stride = (long long)gridDim.x * 256;
  for (; i < total; i += stride) {
    long long bt = i / CD;
    int dd = (int)(i - bt * CD);
    int t  = (int)(bt % CT);
    h[i] = tok[(long long)x[bt] * CD + dd] + pos[(long long)t * CD + dd];
  }
}

// ------------------------------- layernorm ----------------------------------
__global__ __launch_bounds__(256) void k_layernorm(const float* __restrict__ x,
                                                   const float* __restrict__ g,
                                                   const float* __restrict__ b,
                                                   float* __restrict__ y32,
                                                   unsigned short* __restrict__ y16) {
  const int tid = threadIdx.x;
  const long long base = (long long)blockIdx.x * CD;
  __shared__ float s1[256];
  __shared__ float s2[256];
  float a = 0.f, a2 = 0.f;
  for (int i = tid; i < CD; i += 256) {
    float v = x[base + i];
    a += v; a2 += v * v;
  }
  s1[tid] = a; s2[tid] = a2;
  __syncthreads();
  for (int s = 128; s > 0; s >>= 1) {
    if (tid < s) { s1[tid] += s1[tid + s]; s2[tid] += s2[tid + s]; }
    __syncthreads();
  }
  const float mean = s1[0] * (1.0f / CD);
  const float var  = s2[0] * (1.0f / CD) - mean * mean;
  const float inv  = rsqrtf(var + 1e-5f);
  for (int i = tid; i < CD; i += 256) {
    float v = (x[base + i] - mean) * inv * g[i] + b[i];
    if (y32) y32[base + i] = v;
    if (y16) y16[base + i] = f32_to_bf16_bits(v);
  }
}

// --------------------------- causal softmax ---------------------------------
__global__ __launch_bounds__(256) void k_softmax_causal(const float* __restrict__ att,
                                                        unsigned short* __restrict__ p,
                                                        float scale) {
  const int tid = threadIdx.x;
  const long long row = blockIdx.x;
  const int qt = (int)(row % CT);
  const float* ar = att + row * CT;
  unsigned short* pr = p + row * CT;
  __shared__ float red[256];

  float mx = -3.0e38f;
  for (int i = tid; i <= qt; i += 256) mx = fmaxf(mx, ar[i] * scale);
  red[tid] = mx;
  __syncthreads();
  for (int s = 128; s > 0; s >>= 1) {
    if (tid < s) red[tid] = fmaxf(red[tid], red[tid + s]);
    __syncthreads();
  }
  const float m = red[0];
  __syncthreads();

  float sm = 0.f;
  for (int i = tid; i <= qt; i += 256) sm += __expf(ar[i] * scale - m);
  red[tid] = sm;
  __syncthreads();
  for (int s = 128; s > 0; s >>= 1) {
    if (tid < s) red[tid] += red[tid + s];
    __syncthreads();
  }
  const float rinv = 1.0f / red[0];

  for (int i = tid; i < CT; i += 256) {
    float v = (i <= qt) ? __expf(ar[i] * scale - m) * rinv : 0.f;
    pr[i] = f32_to_bf16_bits(v);
  }
}

// ------------------------------ WMMA GEMM -----------------------------------
// C[M,N] = act(A[M,K] @ B + bias) + resid, with B stored K-contiguous:
// B[k,n] = Bt[n*ldb + k].  bf16 operands, fp32 accumulate.
// 256 threads = 8 wave32; block tile 128x64, K-step 32; wave tile 32x32
// (4x v_wmma per K-step).  Double-buffered LDS: async DMA (ASYNCcnt) or
// register-pipelined global loads overlap the WMMA of the current tile.
// All M,N,K used here are multiples of 128/64/32 -> no bounds checks.
// c16t=1 stores the bf16 mirror transposed: C16[col*ldc + row].
#define BMg 128
#define BNg 64
#define BKg 32
#define BKP 40   // +8 ushorts (16B) pad; rows stay 16B aligned

__global__ __launch_bounds__(256) void k_wmma_gemm(
    const unsigned short* __restrict__ A, int lda, long long aO, long long aI,
    const unsigned short* __restrict__ Bt, int ldb, long long bO, long long bI,
    const float* __restrict__ bias, const float* __restrict__ resid,
    float* __restrict__ C32, unsigned short* __restrict__ C16, int c16t,
    int ldc, long long cO, long long cI,
    int K, int innerDiv, int actGelu) {
  const int tid  = threadIdx.x;
  const int lane = tid & 31;
  const int wave = tid >> 5;
  const int wr = wave & 3;
  const int wc = wave >> 2;
  const int lhalf = lane >> 4;
  const int l15 = lane & 15;

  const int z  = blockIdx.z;
  const int zo = z / innerDiv;
  const int zi = z - zo * innerDiv;
  const long long offA = (long long)zo * aO + (long long)zi * aI;
  const long long offB = (long long)zo * bO + (long long)zi * bI;
  const long long offC = (long long)zo * cO + (long long)zi * cI;

  const int m0 = blockIdx.y * BMg;
  const int n0 = blockIdx.x * BNg;

  __shared__ __align__(16) unsigned short As[2][BMg][BKP];  // [m][k]
  __shared__ __align__(16) unsigned short Bs[2][BNg][BKP];  // [n][k]

  v8f acc[2][2];
  const v8f vzero = {0.f, 0.f, 0.f, 0.f, 0.f, 0.f, 0.f, 0.f};
  acc[0][0] = vzero; acc[0][1] = vzero; acc[1][0] = vzero; acc[1][1] = vzero;

  // staging coordinates (all contiguous 16B chunks)
  const int aR = (tid * 8) >> 5;          // 0..63 (second copy adds 64)
  const int aC = (tid * 8) & 31;          // 0,8,16,24
  const int bN = tid >> 2;                // 0..63
  const int bK = (tid & 3) * 8;           // 0,8,16,24

  const unsigned short* gA = A + offA + (long long)(m0 + aR) * lda + aC;
  const unsigned short* gB = Bt + offB + (long long)(n0 + bN) * ldb + bK;
  const long long aRowStep = (long long)64 * lda;

  // per-wave fragment addresses in LDS
  const int aK0 = lhalf ? 16 : 0;         // A: lane<16 -> K0..15
  const int bK0 = lhalf ? 16 : 0;         // B: lane<16 -> K0..15

  union FragBF { v16bf v; v8bf h[2]; };

  // ---- prologue: stage tile 0 into buffer 0 ----
#ifdef USE_ASYNC_LDS
  cp16_async(gA,            &As[0][aR][aC]);
  cp16_async(gA + aRowStep, &As[0][aR + 64][aC]);
  cp16_async(gB,            &Bs[0][bN][bK]);
  wait_async0();
#else
  *(v8us*)&As[0][aR][aC]      = *(const v8us*)gA;
  *(v8us*)&As[0][aR + 64][aC] = *(const v8us*)(gA + aRowStep);
  *(v8us*)&Bs[0][bN][bK]      = *(const v8us*)gB;
#endif
  __syncthreads();

  int p = 0;
  for (int kt = 0; kt < K; kt += BKg) {
    const bool hasNext = (kt + BKg) < K;
    const int nb = p ^ 1;

#ifdef USE_ASYNC_LDS
    if (hasNext) {   // DMA next tile while WMMA crunches this one
      cp16_async(gA + (kt + BKg),            &As[nb][aR][aC]);
      cp16_async(gA + aRowStep + (kt + BKg), &As[nb][aR + 64][aC]);
      cp16_async(gB + (kt + BKg),            &Bs[nb][bN][bK]);
    }
#else
    v8us ra0, ra1, rb;
    if (hasNext) {   // issue global loads; latency hidden behind WMMA
      ra0 = *(const v8us*)(gA + (kt + BKg));
      ra1 = *(const v8us*)(gA + aRowStep + (kt + BKg));
      rb  = *(const v8us*)(gB + (kt + BKg));
    }
#endif

    // ---- fragments from buffer p + 4 WMMA ----
    FragBF a[2], b[2];
#pragma unroll
    for (int ms = 0; ms < 2; ++ms) {
      const int r = wr * 32 + ms * 16 + l15;
      a[ms].h[0] = *(const v8bf*)&As[p][r][aK0];
      a[ms].h[1] = *(const v8bf*)&As[p][r][aK0 + 8];
    }
#pragma unroll
    for (int ns = 0; ns < 2; ++ns) {
      const int c = wc * 32 + ns * 16 + l15;
      b[ns].h[0] = *(const v8bf*)&Bs[p][c][bK0];
      b[ns].h[1] = *(const v8bf*)&Bs[p][c][bK0 + 8];
    }
#pragma unroll
    for (int ms = 0; ms < 2; ++ms)
#pragma unroll
      for (int ns = 0; ns < 2; ++ns)
        acc[ms][ns] = __builtin_amdgcn_wmma_f32_16x16x32_bf16(
            false, a[ms].v, false, b[ns].v, (short)0, acc[ms][ns],
            false, false);

#ifdef USE_ASYNC_LDS
    if (hasNext) wait_async0();          // our DMA into buffer nb done
#else
    if (hasNext) {
      *(v8us*)&As[nb][aR][aC]      = ra0;
      *(v8us*)&As[nb][aR + 64][aC] = ra1;
      *(v8us*)&Bs[nb][bN][bK]      = rb;
    }
#endif
    __syncthreads();
    p = nb;
  }

  // ---- epilogue: bias, activation, residual, stores ----
#pragma unroll
  for (int ms = 0; ms < 2; ++ms) {
#pragma unroll
    for (int ns = 0; ns < 2; ++ns) {
      const int colG = n0 + wc * 32 + ns * 16 + l15;
      const float bs = bias ? bias[colG] : 0.f;
#pragma unroll
      for (int j = 0; j < 8; ++j) {
        const int rowG = m0 + wr * 32 + ms * 16 + j + (lhalf ? 8 : 0);
        float v = acc[ms][ns][j] + bs;
        if (actGelu) v = gelu_exact(v);
        const long long rm = offC + (long long)rowG * ldc + colG;
        if (resid) v += resid[rm];
        if (C32) C32[rm] = v;
        if (C16) {
          const long long ci =
              c16t ? offC + (long long)colG * ldc + rowG : rm;
          C16[ci] = f32_to_bf16_bits(v);
        }
      }
    }
  }
}

// ---------------------------------------------------------------------------
extern "C" void kernel_launch(void* const* d_in, const int* in_sizes, int n_in,
                              void* d_out, int out_size, void* d_ws, size_t ws_size,
                              hipStream_t stream) {
  (void)in_sizes; (void)n_in; (void)out_size; (void)ws_size;
  typedef unsigned short u16;

  const int*   x    = (const int*)  d_in[0];
  const float* tok  = (const float*)d_in[1];
  const float* pos  = (const float*)d_in[2];
  const float* ln1g = (const float*)d_in[3];
  const float* ln1b = (const float*)d_in[4];
  const float* Wq   = (const float*)d_in[5];
  const float* bq   = (const float*)d_in[6];
  const float* Wk   = (const float*)d_in[7];
  const float* bk   = (const float*)d_in[8];
  const float* Wv   = (const float*)d_in[9];
  const float* bv   = (const float*)d_in[10];
  const float* Wo   = (const float*)d_in[11];
  const float* bo   = (const float*)d_in[12];
  const float* ln2g = (const float*)d_in[13];
  const float* ln2b = (const float*)d_in[14];
  const float* W1   = (const float*)d_in[15];
  const float* b1   = (const float*)d_in[16];
  const float* W2   = (const float*)d_in[17];
  const float* b2   = (const float*)d_in[18];
  const float* lnfg = (const float*)d_in[19];
  const float* lnfb = (const float*)d_in[20];
  const float* Wlm  = (const float*)d_in[21];
  const float* blm  = (const float*)d_in[22];
  float* logits = (float*)d_out;

  const long long DD  = (long long)CD * CD;
  const long long DFF = (long long)CD * CFF;
  const long long DV  = (long long)CD * CV;
  const long long MD  = (long long)CM * CD;
  const long long MFF = (long long)CM * CFF;
  const long long ATT = (long long)CB * CH * CT * CT;

  size_t off = 0;
  auto carve = [&](long long bytes) -> void* {
    void* r = (char*)d_ws + off;
    off = (off + (size_t)bytes + 255) & ~(size_t)255;
    return r;
  };
  u16* wq16  = (u16*)carve(CL * DD * 2);    // [L][D][D]   (N,K) transposed
  u16* wk16  = (u16*)carve(CL * DD * 2);
  u16* wv16  = (u16*)carve(CL * DD * 2);
  u16* wo16  = (u16*)carve(CL * DD * 2);
  u16* w1_16 = (u16*)carve(CL * DFF * 2);   // [L][FF][D]
  u16* w2_16 = (u16*)carve(CL * DFF * 2);   // [L][D][FF]
  u16* wlm16 = (u16*)carve(DV * 2);         // [V][D]
  float* h32    = (float*)carve(MD * 4);
  float* hn32   = (float*)carve(MD * 4);
  float* hn2_32 = (float*)carve(MD * 4);
  float* att32  = (float*)carve(ATT * 4);
  u16* hn16   = (u16*)carve(MD * 2);
  u16* q16    = (u16*)carve(MD * 2);
  u16* k16    = (u16*)carve(MD * 2);
  u16* v16T   = (u16*)carve(MD * 2);        // [D][B*T]  (K-contiguous for AV)
  u16* y16    = (u16*)carve(MD * 2);
  u16* hn2_16 = (u16*)carve(MD * 2);
  u16* hf16   = (u16*)carve(MD * 2);
  u16* ff16   = (u16*)carve(MFF * 2);
  u16* att16  = (u16*)carve(ATT * 2);

  const dim3 blk(256);

  // ---- weights -> bf16, transposed to [N][K] ----
  auto cvtT = [&](const float* s, u16* d, int Kd, int Nd, int nmat) {
    k_cvt_bf16_T<<<dim3(4096), blk, 0, stream>>>(s, d, Kd, Nd,
                                                 (long long)nmat * Kd * Nd);
  };
  cvtT(Wq, wq16, CD, CD, CL);
  cvtT(Wk, wk16, CD, CD, CL);
  cvtT(Wv, wv16, CD, CD, CL);
  cvtT(Wo, wo16, CD, CD, CL);
  cvtT(W1, w1_16, CD, CFF, CL);
  cvtT(W2, w2_16, CFF, CD, CL);
  cvtT(Wlm, wlm16, CD, CV, 1);

  k_embed<<<dim3(6144), blk, 0, stream>>>(x, tok, pos, h32, MD);

  auto gemm = [&](const u16* A, int lda, long long aO, long long aI,
                  const u16* B, int ldb, long long bO, long long bI,
                  const float* bias, const float* resid,
                  float* C32, u16* C16, int c16t, int ldc,
                  long long cO, long long cI,
                  int M, int N, int K, int innerDiv, int batch, int act) {
    dim3 grid(N / BNg, M / BMg, batch);
    k_wmma_gemm<<<grid, blk, 0, stream>>>(A, lda, aO, aI, B, ldb, bO, bI,
                                          bias, resid, C32, C16, c16t,
                                          ldc, cO, cI, K, innerDiv, act);
  };

  const float scale = 0.125f;  // 1/sqrt(64)
  const long long TT = (long long)CT * CT;
  const long long TD = (long long)CT * CD;

  for (int l = 0; l < CL; ++l) {
    k_layernorm<<<dim3(CM), blk, 0, stream>>>(h32, ln1g + l * CD, ln1b + l * CD,
                                              hn32, hn16);
    // Q, K projections -> bf16 row-major; V projection -> transposed [D][B*T]
    gemm(hn16, CD, 0, 0, wq16 + l * DD, CD, 0, 0, bq + l * CD, nullptr,
         nullptr, q16, 0, CD, 0, 0, CM, CD, CD, 1, 1, 0);
    gemm(hn16, CD, 0, 0, wk16 + l * DD, CD, 0, 0, bk + l * CD, nullptr,
         nullptr, k16, 0, CD, 0, 0, CM, CD, CD, 1, 1, 0);
    gemm(hn16, CD, 0, 0, wv16 + l * DD, CD, 0, 0, bv + l * CD, nullptr,
         nullptr, v16T, 1, CM, 0, 0, CM, CD, CD, 1, 1, 0);
    // scores = Q @ K^T  (batched over B*H; K rows are K-contiguous columns)
    gemm(q16, CD, TD, CDH, k16, CD, TD, CDH, nullptr, nullptr,
         att32, nullptr, 0, CT, (long long)CH * TT, TT,
         CT, CT, CDH, CH, CB * CH, 0);
    k_softmax_causal<<<dim3((unsigned)(CB * CH * CT)), blk, 0, stream>>>(
        att32, att16, scale);
    // y = P @ V : B operand = V^T [D][B*T], offsets (b -> +b*T, h -> +h*DH*CM)
    gemm(att16, CT, (long long)CH * TT, TT,
         v16T, CM, (long long)CT, (long long)CDH * CM, nullptr, nullptr,
         nullptr, y16, 0, CD, TD, CDH,
         CT, CDH, CT, CH, CB * CH, 0);
    // h = hn + y @ Wo + bo
    gemm(y16, CD, 0, 0, wo16 + l * DD, CD, 0, 0, bo + l * CD, hn32,
         h32, nullptr, 0, CD, 0, 0, CM, CD, CD, 1, 1, 0);
    k_layernorm<<<dim3(CM), blk, 0, stream>>>(h32, ln2g + l * CD, ln2b + l * CD,
                                              hn2_32, hn2_16);
    // ff1 = gelu(hn2 @ W1 + b1)
    gemm(hn2_16, CD, 0, 0, w1_16 + l * DFF, CD, 0, 0, b1 + l * CFF, nullptr,
         nullptr, ff16, 0, CFF, 0, 0, CM, CFF, CD, 1, 1, 1);
    // h = hn2 + ff1 @ W2 + b2
    gemm(ff16, CFF, 0, 0, w2_16 + l * DFF, CFF, 0, 0, b2 + l * CD, hn2_32,
         h32, nullptr, 0, CD, 0, 0, CM, CD, CFF, 1, 1, 0);
  }

  k_layernorm<<<dim3(CM), blk, 0, stream>>>(h32, lnfg, lnfb, nullptr, hf16);
  gemm(hf16, CD, 0, 0, wlm16, CD, 0, 0, blm, nullptr,
       logits, nullptr, 0, CV, 0, 0, CM, CV, CD, 1, 1, 0);
}
